// MacroEncoder_90486370992781
// MI455X (gfx1250) — compile-verified, matrix operand order checked by reading, not run on hardware
//
#include <hip/hip_runtime.h>
#include <hip/hip_bf16.h>

// ---------------------------------------------------------------------------
// GCN macro-encoder for MI455X (gfx1250, wave32).
// GEMMs: v_wmma_f32_16x16x32_bf16 (bf16 inputs, f32 accum).
// Aggregation: hardware f32 global atomics, float4 gathers.
// ---------------------------------------------------------------------------

#define HIDC 256   // IN_CH == HID == 256

typedef __attribute__((ext_vector_type(16))) __bf16 bf16x16;
typedef __attribute__((ext_vector_type(8)))  float  f32x8;

// ---------------------------------------------------------------------------
// Degree / normalization
// ---------------------------------------------------------------------------
__global__ void deg_init_kernel(float* __restrict__ deg, int n) {
    int i = blockIdx.x * blockDim.x + threadIdx.x;
    if (i < n) deg[i] = 1.0f;            // self-loop contributes 1
}

__global__ void deg_edges_kernel(const long long* __restrict__ dst,
                                 float* __restrict__ deg, int e) {
    int i = blockIdx.x * blockDim.x + threadIdx.x;
    if (i < e) unsafeAtomicAdd(&deg[(int)dst[i]], 1.0f);
}

__global__ void deg_rsqrt_kernel(float* __restrict__ deg, int n) {
    int i = blockIdx.x * blockDim.x + threadIdx.x;
    if (i < n) {
        float d = deg[i];
        deg[i] = (d > 0.0f) ? rsqrtf(d) : 0.0f;   // becomes dinv[]
    }
}

// ---------------------------------------------------------------------------
// Repack a 256x256 f32 weight (row = K, col = N) into per-lane bf16 WMMA
// B-fragments. Fragment index = ((kt*256 + n)*2 + half)*16 bf16 values.
// half 0 holds K = {0..7, 16..23} of the 32-wide K tile, half 1 the rest,
// matching the 16-bit A/B operand packing (2 halves per VGPR).
// ---------------------------------------------------------------------------
__global__ void pack_w_kernel(const float* __restrict__ W, __bf16* __restrict__ out) {
    int idx = blockIdx.x * blockDim.x + threadIdx.x;   // 8 kt * 256 n * 2 half
    if (idx >= 8 * 256 * 2) return;
    int hh = idx & 1;
    int n  = (idx >> 1) & 255;
    int kt = idx >> 9;
    __bf16* o = out + (size_t)idx * 16;
#pragma unroll
    for (int j = 0; j < 16; ++j) {
        int kk = kt * 32 + ((j < 8) ? (hh * 8 + j) : (16 + hh * 8 + (j - 8)));
        o[j] = (__bf16)W[(size_t)kk * HIDC + n];
    }
}

// ---------------------------------------------------------------------------
// Dense GEMM: C[nrows,256] = A[nrows,256] @ W  (W pre-packed bf16 fragments).
// 128 threads = 4 waves. Wave w: rows [block*64 + w*16, +16), 64 columns
// selected by blockIdx.y. 4 f32x8 accumulators, K unrolled 8 x 32.
//
// Row guard: A load row is CLAMPED (not branched) — an A row m feeds only
// D row m, and stores are row-guarded, so clamped rows never leak out.
// This keeps the inner loop branch-free (no EXEC save/restore, no zero-fill).
// ---------------------------------------------------------------------------
__global__ __launch_bounds__(128)
void gemm_wmma_kernel(const float* __restrict__ A,
                      const bf16x16* __restrict__ Bp,
                      float* __restrict__ C, int nrows) {
    const int lane    = threadIdx.x & 31;
    const int wave    = threadIdx.x >> 5;
    const int rowBase = blockIdx.x * 64 + wave * 16;
    const int colBase = blockIdx.y * 64;
    const int m       = lane & 15;     // row within tile (A) / col within tile (B)
    const int hh      = lane >> 4;     // K-half select
    int loadRow = rowBase + m;
    if (loadRow >= nrows) loadRow = nrows - 1;   // clamp, don't branch
    const float* arow = A + (size_t)loadRow * HIDC + hh * 8;

    f32x8 acc0 = {}, acc1 = {}, acc2 = {}, acc3 = {};

#pragma unroll
    for (int kt = 0; kt < 8; ++kt) {
        // ---- A fragment: 16 bf16 = K {0..7,16..23} (lanes 0-15) or
        //                            K {8..15,24..31} (lanes 16-31)
        const float* p = arow + kt * 32;
        float4 x0 = *(const float4*)(p);
        float4 x1 = *(const float4*)(p + 4);
        float4 x2 = *(const float4*)(p + 16);
        float4 x3 = *(const float4*)(p + 20);
        bf16x16 a;
        a[0]  = (__bf16)x0.x; a[1]  = (__bf16)x0.y;
        a[2]  = (__bf16)x0.z; a[3]  = (__bf16)x0.w;
        a[4]  = (__bf16)x1.x; a[5]  = (__bf16)x1.y;
        a[6]  = (__bf16)x1.z; a[7]  = (__bf16)x1.w;
        a[8]  = (__bf16)x2.x; a[9]  = (__bf16)x2.y;
        a[10] = (__bf16)x2.z; a[11] = (__bf16)x2.w;
        a[12] = (__bf16)x3.x; a[13] = (__bf16)x3.y;
        a[14] = (__bf16)x3.z; a[15] = (__bf16)x3.w;

        // ---- B fragments: one contiguous 32B load per tile
        const bf16x16* bbase = Bp + ((size_t)kt * HIDC + colBase + m) * 2 + hh;
        bf16x16 b0 = bbase[0];
        bf16x16 b1 = bbase[32];   // +16 columns * 2 halves
        bf16x16 b2 = bbase[64];
        bf16x16 b3 = bbase[96];

        acc0 = __builtin_amdgcn_wmma_f32_16x16x32_bf16(false, a, false, b0,
                                                       (short)0, acc0, false, false);
        acc1 = __builtin_amdgcn_wmma_f32_16x16x32_bf16(false, a, false, b1,
                                                       (short)0, acc1, false, false);
        acc2 = __builtin_amdgcn_wmma_f32_16x16x32_bf16(false, a, false, b2,
                                                       (short)0, acc2, false, false);
        acc3 = __builtin_amdgcn_wmma_f32_16x16x32_bf16(false, a, false, b3,
                                                       (short)0, acc3, false, false);
    }

    // C/D layout: VGPR r -> row r + 8*hh, lane&15 -> column
#pragma unroll
    for (int r = 0; r < 8; ++r) {
        int orow = rowBase + r + hh * 8;
        if (orow < nrows) {
            float* op = C + (size_t)orow * HIDC + colBase + m;
            op[0]  = acc0[r];
            op[16] = acc1[r];
            op[32] = acc2[r];
            op[48] = acc3[r];
        }
    }
}

// ---------------------------------------------------------------------------
// Aggregation: out = dinv[i]^2 * tmp[i]  (self-loop term), float4 streams.
// ---------------------------------------------------------------------------
__global__ void agg_init_kernel(const float* __restrict__ tmp,
                                const float* __restrict__ dinv,
                                float* __restrict__ out, int n) {
    int i = blockIdx.x * blockDim.x + threadIdx.x;   // over n*64 float4s
    if (i >= n * 64) return;
    int row = i >> 6;
    float w = dinv[row] * dinv[row];
    float4 v = ((const float4*)tmp)[i];
    v.x *= w; v.y *= w; v.z *= w; v.w *= w;
    ((float4*)out)[i] = v;
}

// 4 edges per 256-thread block; each lane: float4 gather + 4 f32 atomics.
__global__ __launch_bounds__(256)
void edge_scatter_kernel(const float* __restrict__ tmp,
                         const float* __restrict__ dinv,
                         const long long* __restrict__ src,
                         const long long* __restrict__ dst,
                         float* __restrict__ out, int e) {
    int t = threadIdx.x;
    int eIdx = blockIdx.x * 4 + (t >> 6);
    if (eIdx >= e) return;
    int s = (int)src[eIdx];
    int d = (int)dst[eIdx];
    float w = dinv[s] * dinv[d];
    int c = (t & 63) * 4;
    float4 v = *(const float4*)(tmp + (size_t)s * HIDC + c);
    float* op = out + (size_t)d * HIDC + c;
    unsafeAtomicAdd(op + 0, w * v.x);
    unsafeAtomicAdd(op + 1, w * v.y);
    unsafeAtomicAdd(op + 2, w * v.z);
    unsafeAtomicAdd(op + 3, w * v.w);
}

// ---------------------------------------------------------------------------
// Finalize: h = act(h + bias), float4 streams (64 float4 per row).
// ---------------------------------------------------------------------------
__global__ void bias_act_kernel(float* __restrict__ h, const float* __restrict__ b,
                                int nvec, int doRelu) {
    int i = blockIdx.x * blockDim.x + threadIdx.x;   // over n*64 float4s
    if (i >= nvec) return;
    int c = (i & 63) * 4;
    float4 bb = *(const float4*)(b + c);
    float4 v = ((const float4*)h)[i];
    v.x += bb.x; v.y += bb.y; v.z += bb.z; v.w += bb.w;
    if (doRelu) {
        v.x = fmaxf(v.x, 0.0f); v.y = fmaxf(v.y, 0.0f);
        v.z = fmaxf(v.z, 0.0f); v.w = fmaxf(v.w, 0.0f);
    }
    ((float4*)h)[i] = v;
}

// ---------------------------------------------------------------------------
// Gate: alpha = sigmoid(G1 + gWb + G2 + gUb); out = alpha*ht + (1-alpha)*prev
// float4 streams.
// ---------------------------------------------------------------------------
__device__ __forceinline__ float sigmoidf_fast(float z) {
    return 1.0f / (1.0f + __expf(-z));
}

__global__ void gate_kernel(const float* __restrict__ G1, const float* __restrict__ gWb,
                            const float* __restrict__ G2, const float* __restrict__ gUb,
                            const float* __restrict__ ht, const float* __restrict__ prev,
                            float* __restrict__ out, int nvec) {
    int i = blockIdx.x * blockDim.x + threadIdx.x;   // over n*64 float4s
    if (i >= nvec) return;
    int c = (i & 63) * 4;
    float4 g1 = ((const float4*)G1)[i];
    float4 g2 = ((const float4*)G2)[i];
    float4 bw = *(const float4*)(gWb + c);
    float4 bu = *(const float4*)(gUb + c);
    float4 t  = ((const float4*)ht)[i];
    float4 pv = ((const float4*)prev)[i];
    float4 o;
    float a;
    a = sigmoidf_fast(g1.x + bw.x + g2.x + bu.x); o.x = a * t.x + (1.0f - a) * pv.x;
    a = sigmoidf_fast(g1.y + bw.y + g2.y + bu.y); o.y = a * t.y + (1.0f - a) * pv.y;
    a = sigmoidf_fast(g1.z + bw.z + g2.z + bu.z); o.z = a * t.z + (1.0f - a) * pv.z;
    a = sigmoidf_fast(g1.w + bw.w + g2.w + bu.w); o.w = a * t.w + (1.0f - a) * pv.w;
    ((float4*)out)[i] = o;
}

// ---------------------------------------------------------------------------
// Host-side orchestration
// ---------------------------------------------------------------------------
extern "C" void kernel_launch(void* const* d_in, const int* in_sizes, int n_in,
                              void* d_out, int out_size, void* d_ws, size_t ws_size,
                              hipStream_t stream) {
    const float*     x    = (const float*)d_in[0];
    const long long* ei   = (const long long*)d_in[1];   // int64 [2, E]
    const float*     prev = (const float*)d_in[2];
    const float*     W1   = (const float*)d_in[3];
    const float*     b1   = (const float*)d_in[4];
    const float*     W2   = (const float*)d_in[5];
    const float*     b2   = (const float*)d_in[6];
    const float*     gWw  = (const float*)d_in[7];
    const float*     gWb  = (const float*)d_in[8];
    const float*     gUw  = (const float*)d_in[9];
    const float*     gUb  = (const float*)d_in[10];

    const int n = in_sizes[0] / HIDC;     // 50000
    const int e = in_sizes[1] / 2;        // 800000
    const long long* srcIdx = ei;
    const long long* dstIdx = ei + e;

    // --- workspace carve-up
    char* w = (char*)d_ws;
    float* dinv = (float*)w;
    size_t off = (((size_t)n * 4) + 255) & ~(size_t)255;
    __bf16* wp1 = (__bf16*)(w + off);
    __bf16* wp2 = wp1 + 256 * 256;
    __bf16* wpW = wp2 + 256 * 256;
    __bf16* wpU = wpW + 256 * 256;
    off += (size_t)4 * 256 * 256 * 2;
    float* bufA = (float*)(w + off); off += (size_t)n * HIDC * 4;
    float* bufB = (float*)(w + off);

    float* h_tilde = (float*)d_out;
    float* h_t     = h_tilde + (size_t)n * HIDC;

    const int nvec = n * 64;              // n*256 floats as float4
    const dim3 gemmGrid((n + 63) / 64, 4);
    const int  gemmBlock = 128;

    // 1) degree -> dinv
    deg_init_kernel<<<(n + 255) / 256, 256, 0, stream>>>(dinv, n);
    deg_edges_kernel<<<(e + 255) / 256, 256, 0, stream>>>(dstIdx, dinv, e);
    deg_rsqrt_kernel<<<(n + 255) / 256, 256, 0, stream>>>(dinv, n);

    // 2) pack all four weights to bf16 WMMA fragments
    pack_w_kernel<<<16, 256, 0, stream>>>(W1,  wp1);
    pack_w_kernel<<<16, 256, 0, stream>>>(W2,  wp2);
    pack_w_kernel<<<16, 256, 0, stream>>>(gWw, wpW);
    pack_w_kernel<<<16, 256, 0, stream>>>(gUw, wpU);

    // 3) conv1: t = x @ W1 ; h = relu(agg(t) + b1)
    gemm_wmma_kernel<<<gemmGrid, gemmBlock, 0, stream>>>(x, (const bf16x16*)wp1, bufA, n);
    agg_init_kernel<<<(nvec + 255) / 256, 256, 0, stream>>>(bufA, dinv, bufB, n);
    edge_scatter_kernel<<<(e + 3) / 4, 256, 0, stream>>>(bufA, dinv, srcIdx, dstIdx, bufB, e);
    bias_act_kernel<<<(nvec + 255) / 256, 256, 0, stream>>>(bufB, b1, nvec, 1);

    // 4) conv2: t = h @ W2 ; h_t = agg(t) + b2   (directly into d_out half 2)
    gemm_wmma_kernel<<<gemmGrid, gemmBlock, 0, stream>>>(bufB, (const bf16x16*)wp2, bufA, n);
    agg_init_kernel<<<(nvec + 255) / 256, 256, 0, stream>>>(bufA, dinv, h_t, n);
    edge_scatter_kernel<<<(e + 3) / 4, 256, 0, stream>>>(bufA, dinv, srcIdx, dstIdx, h_t, e);
    bias_act_kernel<<<(nvec + 255) / 256, 256, 0, stream>>>(h_t, b2, nvec, 0);

    // 5) gate: G1 = h_t @ gWw, G2 = prev @ gUw, blend
    gemm_wmma_kernel<<<gemmGrid, gemmBlock, 0, stream>>>(h_t,  (const bf16x16*)wpW, bufA, n);
    gemm_wmma_kernel<<<gemmGrid, gemmBlock, 0, stream>>>(prev, (const bf16x16*)wpU, bufB, n);
    gate_kernel<<<(nvec + 255) / 256, 256, 0, stream>>>(bufA, gWb, bufB, gUb,
                                                        h_t, prev, h_tilde, nvec);
}
